// VitModel_19318762897717
// MI455X (gfx1250) — matrix operand order repaired and checked by reading, not compiled, not run on hardware
//
#include <hip/hip_runtime.h>

// ---------------------------------------------------------------------------
// Types for CDNA5 WMMA (wave32): bf16 A/B fragments (8 VGPRs), f32 C/D (8 VGPRs)
// ---------------------------------------------------------------------------
typedef __attribute__((ext_vector_type(16))) __bf16 bf16x16;
typedef __attribute__((ext_vector_type(8)))  __bf16 bf16x8;
typedef __attribute__((ext_vector_type(8)))  float  f32x8;

#define B_   32
#define IMG_ 512
#define P_   32
#define G_   16
#define NP_  256
#define PD_  3072
#define D_   512
#define H_   8
#define DH_  64
#define LL_  4

__device__ __forceinline__ float gelu_exact(float x) {
    return 0.5f * x * (1.0f + erff(x * 0.70710678118654752f));
}

// ---------------------------------------------------------------------------
// Patch unfold: images [B,512,512,3] f32 -> patches [B*N, 3072] bf16
// flatten order (ph, pw, c), patch order (gy, gx)
// ---------------------------------------------------------------------------
__global__ void patch_kernel(const float* __restrict__ img, __bf16* __restrict__ out) {
    size_t idx = (size_t)blockIdx.x * blockDim.x + threadIdx.x;
    size_t total = (size_t)B_ * NP_ * PD_;
    if (idx >= total) return;
    int j  = (int)(idx % PD_);
    int rp = (int)(idx / PD_);
    int p  = rp % NP_;
    int b  = rp / NP_;
    int c  = j % 3;
    int pw = (j / 3) % P_;
    int ph = j / (3 * P_);
    int gy = p / G_, gx = p % G_;
    size_t src = (((size_t)b * IMG_ + (size_t)(gy * P_ + ph)) * IMG_ + (size_t)(gx * P_ + pw)) * 3 + c;
    out[idx] = (__bf16)img[src];
}

// ---------------------------------------------------------------------------
// LayerNorm over last dim: x f32 [rows, D] -> out bf16 [rows, D]
// ---------------------------------------------------------------------------
__global__ void ln_kernel(const float* __restrict__ x, const float* __restrict__ g,
                          const float* __restrict__ bt, __bf16* __restrict__ out, int D) {
    __shared__ float red[256];
    int row = blockIdx.x;
    int tid = threadIdx.x;
    const float* xr = x + (size_t)row * D;
    float s = 0.f, s2 = 0.f;
    for (int j = tid; j < D; j += 256) { float v = xr[j]; s += v; s2 += v * v; }
    red[tid] = s; __syncthreads();
    for (int off = 128; off > 0; off >>= 1) { if (tid < off) red[tid] += red[tid + off]; __syncthreads(); }
    float mean = red[0] / (float)D; __syncthreads();
    red[tid] = s2; __syncthreads();
    for (int off = 128; off > 0; off >>= 1) { if (tid < off) red[tid] += red[tid + off]; __syncthreads(); }
    float var = red[0] / (float)D - mean * mean;
    float inv = rsqrtf(var + 1e-6f);
    __bf16* orow = out + (size_t)row * D;
    for (int j = tid; j < D; j += 256) {
        float v = (xr[j] - mean) * inv * g[j] + bt[j];
        orow[j] = (__bf16)v;
    }
}

// ---------------------------------------------------------------------------
// Softmax over 256-wide rows with 1/sqrt(DH)=0.125 scale: f32 -> bf16
// ---------------------------------------------------------------------------
__global__ void softmax_kernel(const float* __restrict__ s, __bf16* __restrict__ attn) {
    __shared__ float red[256];
    int row = blockIdx.x;
    int tid = threadIdx.x;
    float v = s[(size_t)row * 256 + tid] * 0.125f;
    red[tid] = v; __syncthreads();
    for (int off = 128; off > 0; off >>= 1) { if (tid < off) red[tid] = fmaxf(red[tid], red[tid + off]); __syncthreads(); }
    float m = red[0]; __syncthreads();
    float e = __expf(v - m);
    red[tid] = e; __syncthreads();
    for (int off = 128; off > 0; off >>= 1) { if (tid < off) red[tid] += red[tid + off]; __syncthreads(); }
    float inv = 1.f / red[0];
    attn[(size_t)row * 256 + tid] = (__bf16)(e * inv);
}

// ---------------------------------------------------------------------------
// Split qkv f32 [B,N,3D] -> q,k bf16 [B,H,N,DH], vT bf16 [B,H,DH,N]
// ---------------------------------------------------------------------------
__global__ void split_qkv_kernel(const float* __restrict__ qkv,
                                 __bf16* __restrict__ q, __bf16* __restrict__ k,
                                 __bf16* __restrict__ vT) {
    size_t idx = (size_t)blockIdx.x * blockDim.x + threadIdx.x;
    size_t total = (size_t)B_ * NP_ * 3 * D_;
    if (idx >= total) return;
    int col = (int)(idx % (3 * D_));
    int rn  = (int)(idx / (3 * D_));
    int n = rn % NP_;
    int b = rn / NP_;
    int which = col / D_;
    int d = col % D_;
    int h = d / DH_, dh = d % DH_;
    __bf16 val = (__bf16)qkv[idx];
    if (which == 0)      q [(((size_t)b * H_ + h) * NP_ + n) * DH_ + dh] = val;
    else if (which == 1) k [(((size_t)b * H_ + h) * NP_ + n) * DH_ + dh] = val;
    else                 vT[(((size_t)b * H_ + h) * DH_ + dh) * NP_ + n] = val;
}

// ---------------------------------------------------------------------------
// Generic bf16 WMMA GEMM:  out = epi( A[M,K]bf16 @ B[K,N] + bias (+resid) )
//   BMODE 0: B source column-major bf16 Bt[N,K]
//   BMODE 1: B source column-major f32  Bt[N,K]
//   BMODE 2: B source row-major    f32  B[K,N]
// Block: 128 threads = 4 waves; block tile 64(M) x 64(N); K step 32.
// Each wave owns a distinct 16-row slice and 4 accumulator tiles across the
// 64 columns (A fragment reused 4x). B staged once per k-step into a shared
// 64x32 bf16 LDS tile -> 32 flops/byte of global traffic per block.
// Batched via blockIdx.z with outer/inner strides (inner count = batchInner).
// ---------------------------------------------------------------------------
template <int BMODE>
__global__ void gemm_wmma(const __bf16* __restrict__ A, long lda,
                          const void* __restrict__ Bv, long ldb,
                          const float* __restrict__ bias,
                          const float* __restrict__ resid, long ldres, int resMod,
                          float* __restrict__ outF, __bf16* __restrict__ outB, long ldc,
                          int M, int Ncols, int K,
                          int batchInner,
                          long aO, long aI, long bO, long bI, long cO, long cI,
                          int actGelu) {
    __shared__ __bf16 ldsB[64 * 32];  // [n_local][k] bf16 tile (4 KB)

    int lane = threadIdx.x & 31;
    int wave = threadIdx.x >> 5;
    int half = lane >> 4;
    int nl   = lane & 15;
    int tileM = blockIdx.y * 64;
    int tileN = blockIdx.x * 64;

    int z  = blockIdx.z;
    int bi = (batchInner > 1) ? (z % batchInner) : 0;
    int bz = (batchInner > 1) ? (z / batchInner) : z;
    const __bf16* Ab = A + (size_t)bz * aO + (size_t)bi * aI;
    size_t bOff = (size_t)bz * bO + (size_t)bi * bI;
    size_t cOff = (size_t)bz * cO + (size_t)bi * cI;

    int rowW = tileM + wave * 16 + nl;            // this lane's A row
    int rowA = (rowW < M) ? rowW : 0;             // clamped for loads
    const __bf16* aRow = Ab + (size_t)rowA * lda;

    f32x8 acc[4];
#pragma unroll
    for (int t = 0; t < 4; ++t) {
        f32x8 zz = {0.f, 0.f, 0.f, 0.f, 0.f, 0.f, 0.f, 0.f};
        acc[t] = zz;
    }

    for (int k0 = 0; k0 < K; k0 += 32) {
        __syncthreads();  // previous LDS tile fully consumed

        // ---- stage B[k0:k0+32, tileN:tileN+64] into ldsB[n][k] as bf16 ----
        if constexpr (BMODE == 0) {
            const __bf16* Bt = (const __bf16*)Bv + bOff;
            int n  = threadIdx.x >> 1;            // 0..63
            int kc = (threadIdx.x & 1) * 16;      // 0 or 16
            bf16x16 v;
            if (tileN + n < Ncols) {
                v = *(const bf16x16*)(Bt + (size_t)(tileN + n) * ldb + k0 + kc);
            } else {
#pragma unroll
                for (int i = 0; i < 16; ++i) v[i] = (__bf16)0.f;
            }
            *(bf16x16*)(&ldsB[n * 32 + kc]) = v;
        } else if constexpr (BMODE == 1) {
            const float* Bt = (const float*)Bv + bOff;
            int n  = threadIdx.x >> 1;
            int kc = (threadIdx.x & 1) * 16;
            bf16x16 v;
            if (tileN + n < Ncols) {
                const float* bp = Bt + (size_t)(tileN + n) * ldb + k0 + kc;
                f32x8 b0 = *(const f32x8*)bp;
                f32x8 b1 = *(const f32x8*)(bp + 8);
#pragma unroll
                for (int i = 0; i < 8; ++i) { v[i] = (__bf16)b0[i]; v[i + 8] = (__bf16)b1[i]; }
            } else {
#pragma unroll
                for (int i = 0; i < 16; ++i) v[i] = (__bf16)0.f;
            }
            *(bf16x16*)(&ldsB[n * 32 + kc]) = v;
        } else {
            const float* Bm = (const float*)Bv + bOff;
            int kk = threadIdx.x >> 2;            // 0..31 : K row within step
            int nc = (threadIdx.x & 3) * 16;      // 0,16,32,48 : N sub-block
            const float* src = Bm + (size_t)(k0 + kk) * ldb + tileN + nc;
            if (k0 + 32 < K) __builtin_prefetch(src + (size_t)32 * ldb, 0, 0);
#pragma unroll
            for (int j = 0; j < 16; ++j) {
                int n = tileN + nc + j;
                float v = (n < Ncols) ? src[j] : 0.f;
                ldsB[(nc + j) * 32 + kk] = (__bf16)v;  // transpose to [n][k]
            }
        }
        __syncthreads();

        // ---- A fragment (16x32 bf16): two 16B chunks per lane ----
        bf16x8 a0 = *(const bf16x8*)(aRow + k0 + 16 * half);
        bf16x8 a1 = *(const bf16x8*)(aRow + k0 + 16 * half + 8);
        bf16x16 af;
#pragma unroll
        for (int i = 0; i < 8; ++i) { af[i] = a0[i]; af[i + 8] = a1[i]; }

        // ---- 4 WMMAs, A fragment reused across the 4 column tiles ----
#pragma unroll
        for (int t = 0; t < 4; ++t) {
            bf16x16 bfr = *(const bf16x16*)(&ldsB[(t * 16 + nl) * 32 + 16 * half]);
            acc[t] = __builtin_amdgcn_wmma_f32_16x16x32_bf16(
                /*neg_a=*/false, af, /*neg_b=*/false, bfr,
                /*c_mod=*/(short)0, acc[t], /*reuse_a=*/false, /*reuse_b=*/false);
        }
    }

    // ---- epilogue: bias -> gelu -> residual -> store (f32 or bf16) ----
#pragma unroll
    for (int t = 0; t < 4; ++t) {
        int colW = tileN + t * 16 + nl;
        if (colW >= Ncols) continue;
        float bv = bias ? bias[colW] : 0.f;
#pragma unroll
        for (int r = 0; r < 8; ++r) {
            int row = tileM + wave * 16 + r + 8 * half;
            if (row >= M) continue;
            float v = acc[t][r] + bv;
            if (actGelu) v = gelu_exact(v);
            if (resid) {
                int rr = resMod ? (row % resMod) : row;
                v += resid[(size_t)rr * ldres + colW];
            }
            size_t oi = cOff + (size_t)row * ldc + colW;
            if (outF) outF[oi] = v;
            else      outB[oi] = (__bf16)v;
        }
    }
}

// ---------------------------------------------------------------------------
// Host-side orchestration
// ---------------------------------------------------------------------------
extern "C" void kernel_launch(void* const* d_in, const int* in_sizes, int n_in,
                              void* d_out, int out_size, void* d_ws, size_t ws_size,
                              hipStream_t stream) {
    (void)in_sizes; (void)n_in; (void)out_size; (void)ws_size;

    const float* images  = (const float*)d_in[0];
    const float* W_proj  = (const float*)d_in[1];
    const float* b_proj  = (const float*)d_in[2];
    const float* pos_emb = (const float*)d_in[3];
    const float* ln1_g   = (const float*)d_in[4];
    const float* ln1_b   = (const float*)d_in[5];
    const float* qkv_w   = (const float*)d_in[6];
    const float* qkv_b   = (const float*)d_in[7];
    const float* aow     = (const float*)d_in[8];
    const float* aob     = (const float*)d_in[9];
    const float* ln2_g   = (const float*)d_in[10];
    const float* ln2_b   = (const float*)d_in[11];
    const float* w1      = (const float*)d_in[12];
    const float* b1      = (const float*)d_in[13];
    const float* w2      = (const float*)d_in[14];
    const float* b2      = (const float*)d_in[15];
    const float* lnf_g   = (const float*)d_in[16];
    const float* lnf_b   = (const float*)d_in[17];
    const float* hW[5]   = {(const float*)d_in[18], (const float*)d_in[20], (const float*)d_in[22],
                            (const float*)d_in[24], (const float*)d_in[26]};
    const float* hb[5]   = {(const float*)d_in[19], (const float*)d_in[21], (const float*)d_in[23],
                            (const float*)d_in[25], (const float*)d_in[27]};
    const float* bbW     = (const float*)d_in[28];
    const float* bbb     = (const float*)d_in[29];
    float* out           = (float*)d_out;

    const int ROWS = B_ * NP_;  // 8192

    // workspace carve-outs (256B aligned)
    char* ws = (char*)d_ws;
    size_t off = 0;
    auto carve = [&](size_t bytes) -> char* {
        char* p = ws + off;
        off = (off + bytes + 255) & ~(size_t)255;
        return p;
    };
    __bf16* patches = (__bf16*)carve((size_t)ROWS * PD_ * 2);       // 50 MB
    float*  x       = (float*) carve((size_t)ROWS * D_ * 4);        // 16 MB
    __bf16* x1      = (__bf16*)carve((size_t)ROWS * D_ * 2);        //  8 MB
    float*  qkv     = (float*) carve((size_t)ROWS * 3 * D_ * 4);    // 50 MB
    __bf16* qb      = (__bf16*)carve((size_t)ROWS * D_ * 2);
    __bf16* kb      = (__bf16*)carve((size_t)ROWS * D_ * 2);
    __bf16* vT      = (__bf16*)carve((size_t)ROWS * D_ * 2);
    float*  scores  = (float*) carve((size_t)B_ * H_ * NP_ * NP_ * 4);  // 67 MB
    __bf16* attn    = (__bf16*)carve((size_t)B_ * H_ * NP_ * NP_ * 2);  // 33 MB
    __bf16* ob      = (__bf16*)carve((size_t)ROWS * D_ * 2);
    float*  x2      = (float*) carve((size_t)ROWS * D_ * 4);
    __bf16* x3      = (__bf16*)carve((size_t)ROWS * D_ * 2);
    __bf16* h1      = (__bf16*)carve((size_t)ROWS * 2 * D_ * 2);    // 16 MB
    __bf16* rep     = (__bf16*)carve((size_t)ROWS * D_ * 2);
    __bf16* f1      = (__bf16*)carve((size_t)B_ * 2048 * 2);
    __bf16* f2      = (__bf16*)carve((size_t)B_ * 1024 * 2);
    __bf16* f3      = (__bf16*)carve((size_t)B_ * 512 * 2);
    __bf16* f4      = (__bf16*)carve((size_t)B_ * 64 * 2);
    __bf16* f5      = (__bf16*)carve((size_t)B_ * 32 * 2);

    auto gemm = [&](int bmode, const __bf16* A, long lda, const void* Bv, long ldb,
                    const float* bias, const float* resid, long ldres, int resMod,
                    float* oF, __bf16* oB, long ldc, int M, int N, int K,
                    int batches, int bInner,
                    long aO, long aI, long bO, long bI, long cO, long cI, int gelu) {
        dim3 g((unsigned)((N + 63) / 64), (unsigned)((M + 63) / 64), (unsigned)batches), blk(128);
        if (bmode == 0)
            gemm_wmma<0><<<g, blk, 0, stream>>>(A, lda, Bv, ldb, bias, resid, ldres, resMod,
                                                oF, oB, ldc, M, N, K, bInner, aO, aI, bO, bI, cO, cI, gelu);
        else if (bmode == 1)
            gemm_wmma<1><<<g, blk, 0, stream>>>(A, lda, Bv, ldb, bias, resid, ldres, resMod,
                                                oF, oB, ldc, M, N, K, bInner, aO, aI, bO, bI, cO, cI, gelu);
        else
            gemm_wmma<2><<<g, blk, 0, stream>>>(A, lda, Bv, ldb, bias, resid, ldres, resMod,
                                                oF, oB, ldc, M, N, K, bInner, aO, aI, bO, bI, cO, cI, gelu);
    };

    // 1) patch unfold
    {
        size_t total = (size_t)ROWS * PD_;
        patch_kernel<<<(unsigned)((total + 255) / 256), 256, 0, stream>>>(images, patches);
    }
    // 2) patch projection: x = patches @ W_proj + b_proj + pos_emb  (pos via row % 256)
    gemm(2, patches, PD_, W_proj, D_, b_proj, pos_emb, D_, NP_,
         x, nullptr, D_, ROWS, D_, PD_, 1, 1, 0, 0, 0, 0, 0, 0, 0);

    // 3) transformer blocks
    for (int l = 0; l < LL_; ++l) {
        ln_kernel<<<ROWS, 256, 0, stream>>>(x, ln1_g + l * D_, ln1_b + l * D_, x1, D_);

        // qkv = x1 @ qkv_w.T + qkv_b : qkv_w [3D,D] is column-major B => BMODE1
        gemm(1, x1, D_, qkv_w + (size_t)l * 3 * D_ * D_, D_, qkv_b + l * 3 * D_,
             nullptr, 0, 0, qkv, nullptr, 3 * D_, ROWS, 3 * D_, D_,
             1, 1, 0, 0, 0, 0, 0, 0, 0);

        {
            size_t total = (size_t)ROWS * 3 * D_;
            split_qkv_kernel<<<(unsigned)((total + 255) / 256), 256, 0, stream>>>(qkv, qb, kb, vT);
        }

        // scores = q @ k^T, batched over B*H (Bt = k, naturally [N,DH])
        gemm(0, qb, DH_, kb, DH_, nullptr, nullptr, 0, 0,
             scores, nullptr, NP_, NP_, NP_, DH_,
             B_ * H_, 1,
             (long)NP_ * DH_, 0, (long)NP_ * DH_, 0, (long)NP_ * NP_, 0, 0);

        softmax_kernel<<<B_ * H_ * NP_, 256, 0, stream>>>(scores, attn);

        // o = attn @ v, batched; result written merged-head as bf16 [B,N,D]
        gemm(0, attn, NP_, vT, NP_, nullptr, nullptr, 0, 0,
             nullptr, ob, D_, NP_, DH_, NP_,
             B_ * H_, H_,
             (long)H_ * NP_ * NP_, (long)NP_ * NP_,
             (long)H_ * DH_ * NP_, (long)DH_ * NP_,
             (long)NP_ * D_, (long)DH_, 0);

        // x2 = o @ out_w.T + out_b + x : out_w [D,D] column-major B => BMODE1
        gemm(1, ob, D_, aow + (size_t)l * D_ * D_, D_, aob + l * D_,
             x, D_, 0, x2, nullptr, D_, ROWS, D_, D_,
             1, 1, 0, 0, 0, 0, 0, 0, 0);

        ln_kernel<<<ROWS, 256, 0, stream>>>(x2, ln2_g + l * D_, ln2_b + l * D_, x3, D_);

        // h1 = gelu(x3 @ w1 + b1) : w1 [D,2D] row-major => BMODE2
        gemm(2, x3, D_, w1 + (size_t)l * D_ * 2 * D_, 2 * D_, b1 + l * 2 * D_,
             nullptr, 0, 0, nullptr, h1, 2 * D_, ROWS, 2 * D_, D_,
             1, 1, 0, 0, 0, 0, 0, 0, 1);

        // x = gelu(h1 @ w2 + b2) + x2 : w2 [2D,D] row-major => BMODE2
        gemm(2, h1, 2 * D_, w2 + (size_t)l * 2 * D_ * D_, D_, b2 + l * D_,
             x2, D_, 0, x, nullptr, D_, ROWS, D_, 2 * D_,
             1, 1, 0, 0, 0, 0, 0, 0, 1);
    }

    // 4) final LN -> rep bf16 [B, N*D] (== [B*N, D] contiguous)
    ln_kernel<<<ROWS, 256, 0, stream>>>(x, lnf_g, lnf_b, rep, D_);

    // 5) MLP head: M=32 tall-skinny GEMMs, weights read as f32 row-major (BMODE2)
    const int HD[6] = {NP_ * D_, 2048, 1024, 512, 64, 32};
    __bf16* fbuf[6] = {rep, f1, f2, f3, f4, f5};
    for (int i = 0; i < 5; ++i) {
        gemm(2, fbuf[i], HD[i], hW[i], HD[i + 1], hb[i],
             nullptr, 0, 0, nullptr, fbuf[i + 1], HD[i + 1],
             B_, HD[i + 1], HD[i], 1, 1, 0, 0, 0, 0, 0, 0, 1);
    }
    // 6) bbox: out = f5 @ bbW + bbb   (N=4 handled by guards)
    gemm(2, f5, 32, bbW, 4, bbb, nullptr, 0, 0,
         out, nullptr, 4, B_, 4, 32, 1, 1, 0, 0, 0, 0, 0, 0, 0);
}
